// TopKSAE_8727373546165
// MI455X (gfx1250) — matrix operand order repaired and checked by reading, not compiled, not run on hardware
//
#include <hip/hip_runtime.h>
#include <hip/hip_bf16.h>

#define BATCH   8192
#define D_MODEL 768
#define FDIM    24576
#define TOPK    128

typedef __attribute__((ext_vector_type(16))) __bf16 v16bf;
typedef __attribute__((ext_vector_type(8)))  __bf16 v8bf;
typedef __attribute__((ext_vector_type(8)))  float  v8f;

// ---------------- f32 -> bf16 (RNE) conversion ----------------
__device__ __forceinline__ unsigned short f2bf(float f) {
    unsigned int u = __float_as_uint(f);
    unsigned int r = u + 0x7FFFu + ((u >> 16) & 1u);
    return (unsigned short)(r >> 16);
}

__global__ __launch_bounds__(256) void cvt_bf16_kernel(const float* __restrict__ in,
                                                       unsigned short* __restrict__ out,
                                                       int n) {
    int i = (blockIdx.x * 256 + threadIdx.x) * 4;
    if (i + 3 < n) {
        float4 v = *(const float4*)(in + i);
        out[i + 0] = f2bf(v.x);
        out[i + 1] = f2bf(v.y);
        out[i + 2] = f2bf(v.z);
        out[i + 3] = f2bf(v.w);
    }
}

// ---------------- W_dec [768, 24576] -> W_decT [24576, 768] ----------------
__global__ __launch_bounds__(256) void transpose_wdec_kernel(const float* __restrict__ wd,
                                                             float* __restrict__ wdT) {
    __shared__ float tile[32][33];
    int x = blockIdx.x * 32 + threadIdx.x;   // feature col in wd
    int y = blockIdx.y * 32 + threadIdx.y;   // d row in wd
    #pragma unroll
    for (int j = 0; j < 32; j += 8)
        tile[threadIdx.y + j][threadIdx.x] = wd[(size_t)(y + j) * FDIM + x];
    __syncthreads();
    int tx = blockIdx.y * 32 + threadIdx.x;  // d
    int ty = blockIdx.x * 32 + threadIdx.y;  // feature
    #pragma unroll
    for (int j = 0; j < 32; j += 8)
        wdT[(size_t)(ty + j) * D_MODEL + tx] = tile[threadIdx.x][threadIdx.y + j];
}

// ---------------- Encode GEMM: pre_acts = x @ W_enc^T + b_enc ----------------
// Wave tile: M=32, N=64.  Block = 8 waves -> M=32, N=512.
// A (x) 16x32 bf16 frag: lanes 0-15 hold M=0..15, K=kbase..kbase+7 (V0-3) and
// K=kbase+16..+23 (V4-7), kbase = (lane>>4)*8.
// B (W_enc row-major N x K) 32x16 frag: lane n=lane&15, K=kb..kb+7 and
// K=kb+16..kb+23, kb = kk + (lane>>4)*8.
__global__ __launch_bounds__(256) void encode_gemm_kernel(const unsigned short* __restrict__ xb,
                                                          const unsigned short* __restrict__ wb,
                                                          const float* __restrict__ b_enc,
                                                          float* __restrict__ z) {
    const int lane  = threadIdx.x & 31;
    const int wave  = threadIdx.x >> 5;
    const int l15   = lane & 15;
    const int lhalf = lane >> 4;

    const int m0 = blockIdx.y * 32;
    const int n0 = blockIdx.x * 512 + wave * 64;

    v8f acc[2][4];
    #pragma unroll
    for (int mh = 0; mh < 2; ++mh)
        #pragma unroll
        for (int ns = 0; ns < 4; ++ns)
            acc[mh][ns] = (v8f){0.f, 0.f, 0.f, 0.f, 0.f, 0.f, 0.f, 0.f};

    const unsigned short* xrow0 = xb + (size_t)(m0 + l15) * D_MODEL;
    const unsigned short* xrow1 = xrow0 + (size_t)16 * D_MODEL;

    for (int kk = 0; kk < D_MODEL; kk += 32) {
        const int ka = kk + lhalf * 8;
        v16bf aF[2];
        {
            v8bf lo = *(const v8bf*)(xrow0 + ka);
            v8bf hi = *(const v8bf*)(xrow0 + ka + 16);
            aF[0] = __builtin_shufflevector(lo, hi, 0,1,2,3,4,5,6,7,8,9,10,11,12,13,14,15);
            lo = *(const v8bf*)(xrow1 + ka);
            hi = *(const v8bf*)(xrow1 + ka + 16);
            aF[1] = __builtin_shufflevector(lo, hi, 0,1,2,3,4,5,6,7,8,9,10,11,12,13,14,15);
        }
        const int kb = kk + lhalf * 8;
        #pragma unroll
        for (int ns = 0; ns < 4; ++ns) {
            const unsigned short* wrow = wb + (size_t)(n0 + ns * 16 + l15) * D_MODEL + kb;
            v8bf b0 = *(const v8bf*)(wrow);
            v8bf b1 = *(const v8bf*)(wrow + 16);
            v16bf bF = __builtin_shufflevector(b0, b1, 0,1,2,3,4,5,6,7,8,9,10,11,12,13,14,15);
            acc[0][ns] = __builtin_amdgcn_wmma_f32_16x16x32_bf16(
                false, aF[0], false, bF, (short)0, acc[0][ns], false, false);
            acc[1][ns] = __builtin_amdgcn_wmma_f32_16x16x32_bf16(
                false, aF[1], false, bF, (short)0, acc[1][ns], false, false);
        }
    }

    // Epilogue: add bias, store pre_acts into z region (dense f32).
    // C/D layout: VGPR r, lanes 0-15 -> (M=r, N=lane); lanes 16-31 -> (M=r+8, N=lane-16).
    #pragma unroll
    for (int ns = 0; ns < 4; ++ns) {
        const int col = n0 + ns * 16 + l15;
        const float bias = b_enc[col];
        #pragma unroll
        for (int mh = 0; mh < 2; ++mh) {
            const int rbase = m0 + mh * 16 + lhalf * 8;
            float* zp = z + (size_t)rbase * FDIM + col;
            #pragma unroll
            for (int r = 0; r < 8; ++r)
                zp[(size_t)r * FDIM] = acc[mh][ns][r] + bias;
        }
    }
}

// ---------------- TopK (exact 128th-largest via 4-pass radix select) ----------------
__device__ __forceinline__ unsigned int key_of(float f) {
    unsigned int u = __float_as_uint(f);
    return (u & 0x80000000u) ? ~u : (u | 0x80000000u);  // monotone transform
}

__global__ __launch_bounds__(256) void topk_kernel(float* __restrict__ z,
                                                   int* __restrict__ idx_out,
                                                   float* __restrict__ val_out,
                                                   int* __restrict__ cnt_out) {
    extern __shared__ float rowf[];            // FDIM floats (96 KB dynamic LDS)
    __shared__ unsigned int hist[256];
    __shared__ unsigned int s_prefix, s_remain, s_cnt;

    const int row = blockIdx.x;
    const int tid = threadIdx.x;
    float* zrow = z + (size_t)row * FDIM;

    for (int j = tid; j < FDIM; j += 256) rowf[j] = zrow[j];
    __syncthreads();

    unsigned int prefix = 0u, pmask = 0u, remain = TOPK;
    for (int level = 0; level < 4; ++level) {
        const int shift = 24 - 8 * level;
        hist[tid] = 0u;
        __syncthreads();
        for (int j = tid; j < FDIM; j += 256) {
            unsigned int k = key_of(rowf[j]);
            if ((k & pmask) == prefix)
                atomicAdd(&hist[(k >> shift) & 255u], 1u);
        }
        __syncthreads();
        if (tid == 0) {
            unsigned int cum = 0u;
            for (int b = 255; b >= 0; --b) {
                unsigned int c = hist[b];
                if (cum + c >= remain) {
                    s_prefix = prefix | ((unsigned int)b << shift);
                    s_remain = remain - cum;
                    break;
                }
                cum += c;
            }
        }
        __syncthreads();
        prefix = s_prefix;
        remain = s_remain;
        pmask |= (0xFFu << shift);
        __syncthreads();
    }
    const unsigned int thresh = prefix;        // key of the 128th largest

    if (tid == 0) s_cnt = 0u;
    __syncthreads();

    // z[j] = v if (v >= T && v > 0) else 0; compact positives into (idx, val).
    for (int j = tid; j < FDIM; j += 256) {
        float f = rowf[j];
        unsigned int k = key_of(f);
        bool sel = (k >= thresh) && (k > 0x80000000u);   // > 0x80000000 <=> f > 0
        zrow[j] = sel ? f : 0.0f;
        if (sel) {
            unsigned int p = atomicAdd(&s_cnt, 1u);
            if (p < TOPK) {
                idx_out[(size_t)row * TOPK + p] = j;
                val_out[(size_t)row * TOPK + p] = f;
            }
        }
    }
    __syncthreads();
    if (tid == 0) cnt_out[row] = (int)(s_cnt < (unsigned)TOPK ? s_cnt : (unsigned)TOPK);
}

// ---------------- Sparse decode: recon[b,:] = sum_k val_k * W_decT[idx_k,:] ----------------
__global__ __launch_bounds__(256) void decode_kernel(const float* __restrict__ wdT,
                                                     const int* __restrict__ idxs,
                                                     const float* __restrict__ vals,
                                                     const int* __restrict__ cnts,
                                                     float* __restrict__ recon) {
    const int row = blockIdx.x;
    const int tid = threadIdx.x;
    const int cnt = cnts[row];
    const int* ip = idxs + (size_t)row * TOPK;
    const float* vp = vals + (size_t)row * TOPK;

    float a0 = 0.f, a1 = 0.f, a2 = 0.f;
    for (int k = 0; k < cnt; ++k) {
        const int f = ip[k];
        const float v = vp[k];
        const float* wr = wdT + (size_t)f * D_MODEL;
        a0 += v * wr[tid];
        a1 += v * wr[tid + 256];
        a2 += v * wr[tid + 512];
    }
    float* rr = recon + (size_t)row * D_MODEL;
    rr[tid]       = a0;
    rr[tid + 256] = a1;
    rr[tid + 512] = a2;
}

// ---------------- launch ----------------
extern "C" void kernel_launch(void* const* d_in, const int* in_sizes, int n_in,
                              void* d_out, int out_size, void* d_ws, size_t ws_size,
                              hipStream_t stream) {
    const float* x     = (const float*)d_in[0];   // [8192, 768]
    const float* W_enc = (const float*)d_in[1];   // [24576, 768]
    const float* b_enc = (const float*)d_in[2];   // [24576]
    const float* W_dec = (const float*)d_in[3];   // [768, 24576]

    float* recon = (float*)d_out;                               // [8192, 768]
    float* z     = (float*)d_out + (size_t)BATCH * D_MODEL;     // [8192, 24576]

    char* ws = (char*)d_ws;
    size_t off = 0;
    unsigned short* xb  = (unsigned short*)(ws + off); off += (size_t)BATCH * D_MODEL * 2;
    unsigned short* wb  = (unsigned short*)(ws + off); off += (size_t)FDIM * D_MODEL * 2;
    float*          wdT = (float*)(ws + off);          off += (size_t)FDIM * D_MODEL * 4;
    int*            idxs = (int*)(ws + off);           off += (size_t)BATCH * TOPK * 4;
    float*          vals = (float*)(ws + off);         off += (size_t)BATCH * TOPK * 4;
    int*            cnts = (int*)(ws + off);           off += (size_t)BATCH * 4;

    const int n_x = BATCH * D_MODEL;     // 6,291,456 (div by 1024)
    const int n_w = FDIM * D_MODEL;      // 18,874,368 (div by 1024)
    cvt_bf16_kernel<<<n_x / 1024, 256, 0, stream>>>(x, xb, n_x);
    cvt_bf16_kernel<<<n_w / 1024, 256, 0, stream>>>(W_enc, wb, n_w);

    transpose_wdec_kernel<<<dim3(FDIM / 32, D_MODEL / 32), dim3(32, 8), 0, stream>>>(W_dec, wdT);

    encode_gemm_kernel<<<dim3(FDIM / 512, BATCH / 32), 256, 0, stream>>>(xb, wb, b_enc, z);

    topk_kernel<<<BATCH, 256, FDIM * sizeof(float), stream>>>(z, idxs, vals, cnts);

    decode_kernel<<<BATCH, 256, 0, stream>>>(wdT, idxs, vals, cnts, recon);
}
